// LeWinTransformerBlock_39204461478294
// MI455X (gfx1250) — matrix operand order repaired and checked by reading, not compiled, not run on hardware
//
#include <hip/hip_runtime.h>
#include <cmath>

typedef _Float16 f16;
typedef _Float16 v8h  __attribute__((ext_vector_type(8)));
typedef _Float16 v16h __attribute__((ext_vector_type(16)));
typedef float    v8f  __attribute__((ext_vector_type(8)));

#define B_     4
#define H_     128
#define W_     128
#define C_     256
#define HD_    32
#define HID_   1024
#define TOK_   65536
#define SHIFT_ 4

__device__ __forceinline__ v16h cat8(v8h a, v8h b) {
  return __builtin_shufflevector(a, b, 0, 1, 2, 3, 4, 5, 6, 7, 8, 9, 10, 11, 12, 13, 14, 15);
}
// convert-or-copy 8 contiguous elements -> v8h
__device__ __forceinline__ v8h load8(const f16* p) { return *(const v8h*)p; }
__device__ __forceinline__ v8h load8(const float* p) {
  v8h r;
#pragma unroll
  for (int j = 0; j < 8; ++j) r[j] = (f16)p[j];
  return r;
}
// A-fragment (16-bit A 16x32): lane holds contiguous K = hi*16 .. hi*16+15
__device__ __forceinline__ v16h load_afrag(const f16* row, int hi) {
  return cat8(*(const v8h*)(row + hi * 16), *(const v8h*)(row + hi * 16 + 8));
}
// B-fragment (32x16): lane holds contiguous K = hi*16 .. hi*16+15
__device__ __forceinline__ v16h load_bfrag(const f16* row, int hi) {
  return cat8(*(const v8h*)(row + hi * 16), *(const v8h*)(row + hi * 16 + 8));
}
// single-instruction lane-XOR exchange: ds_swizzle_b32, group-of-32 bit-mask mode
// offset = {1'b0, xor[14:10], or[9:5]=0, and[4:0]=0x1f}  ->  src_lane = lane ^ XMASK
template<int XMASK>
__device__ __forceinline__ float xor_swz(float v) {
  return __int_as_float(
      __builtin_amdgcn_ds_swizzle(__float_as_int(v), (XMASK << 10) | 0x1f));
}
__device__ __forceinline__ float gelu_f(float x) {
  return 0.5f * x * (1.0f + erff(x * 0.70710678118654752f));
}
// Swin shift-mask region id for window wi (0..255) and in-window pos p (0..63)
__device__ __forceinline__ int swin_region(int wi, int p) {
  int h = ((wi >> 4) << 3) + (p >> 3);
  int w = ((wi & 15) << 3) + (p & 7);
  int rh = (h < H_ - 8) ? 0 : ((h < H_ - SHIFT_) ? 1 : 2);
  int rw = (w < W_ - 8) ? 0 : ((w < W_ - SHIFT_) ? 1 : 2);
  return rh * 3 + rw;
}

// -------- 1) LayerNorm + roll(-4,-4) + 8x8 window partition -> f16 --------
__global__ void ln_window_kernel(const float* __restrict__ x,
                                 const float* __restrict__ g,
                                 const float* __restrict__ bta,
                                 f16* __restrict__ xw) {
  int lane = threadIdx.x & 31;
  int wave = threadIdx.x >> 5;
  int t = blockIdx.x * 8 + wave;            // destination token (window-ordered)
  int wg = t >> 6, p = t & 63;
  int b = wg >> 8, wi = wg & 255;
  int h2 = ((wi >> 4) << 3) + (p >> 3);
  int w2 = ((wi & 15) << 3) + (p & 7);
  int hs = (h2 + SHIFT_) & (H_ - 1);        // roll(-SHIFT): src = dst + SHIFT
  int ws = (w2 + SHIFT_) & (W_ - 1);
  size_t src = (((size_t)b << 14) + (size_t)hs * W_ + ws) * (size_t)C_;
  float v[8]; float s = 0.f, ss = 0.f;
#pragma unroll
  for (int j = 0; j < 8; ++j) {
    v[j] = x[src + lane * 8 + j];
    s += v[j]; ss += v[j] * v[j];
  }
  s += xor_swz<1>(s);   ss += xor_swz<1>(ss);
  s += xor_swz<2>(s);   ss += xor_swz<2>(ss);
  s += xor_swz<4>(s);   ss += xor_swz<4>(ss);
  s += xor_swz<8>(s);   ss += xor_swz<8>(ss);
  s += xor_swz<16>(s);  ss += xor_swz<16>(ss);
  float mu  = s * (1.0f / C_);
  float var = ss * (1.0f / C_) - mu * mu;
  float inv = rsqrtf(var + 1e-5f);
  v8h outv;
#pragma unroll
  for (int j = 0; j < 8; ++j) {
    int c = lane * 8 + j;
    outv[j] = (f16)((v[j] - mu) * inv * g[c] + bta[c]);
  }
  *(v8h*)(xw + (size_t)t * C_ + lane * 8) = outv;
}

// -------- 2) Generic WMMA GEMM: D[M,N] = A[M,K] * B[N,K]^T (+epilogue) --------
// EPI 0: store f16, (acc + bias) * scale
// EPI 1: proj: window-reverse + unshift scatter, outF[tok] = aux[tok] + acc + bias
// EPI 2: final: outF = aux + gelu(acc)
template<typename TA, int EPI>
__global__ void gemm_wmma_kernel(const TA* __restrict__ A, const float* __restrict__ Bw,
                                 const float* __restrict__ bias,
                                 f16* __restrict__ outH, float* __restrict__ outF,
                                 const float* __restrict__ aux,
                                 int M, int N, int K, float scale) {
  __shared__ f16 As[64][40];
  __shared__ f16 Bs[128][40];
  int nbn = N >> 7;
  int bm = blockIdx.x / nbn;
  int bn = blockIdx.x % nbn;
  int tid = threadIdx.x;
  int lane = tid & 31, wave = tid >> 5;
  int wm = wave >> 2, wn = wave & 3;        // 2 (M) x 4 (N) wave grid
  int r = lane & 15, hi = lane >> 4;

  v8f acc[2][2];
#pragma unroll
  for (int a0 = 0; a0 < 2; ++a0)
#pragma unroll
    for (int a1 = 0; a1 < 2; ++a1)
#pragma unroll
      for (int e = 0; e < 8; ++e) acc[a0][a1][e] = 0.f;

  int arow = tid >> 2, ak = (tid & 3) * 8;
  int brow = tid >> 1, bk = (tid & 1) * 16;
  const TA*    Ap = A  + (size_t)(bm * 64  + arow) * K + ak;
  const float* Bp = Bw + (size_t)(bn * 128 + brow) * K + bk;

  for (int k0 = 0; k0 < K; k0 += 32) {
    *(v8h*)&As[arow][ak]     = load8(Ap + k0);
    *(v8h*)&Bs[brow][bk]     = load8(Bp + k0);
    *(v8h*)&Bs[brow][bk + 8] = load8(Bp + k0 + 8);
    __syncthreads();
    v16h af[2], bf[2];
#pragma unroll
    for (int mt = 0; mt < 2; ++mt) af[mt] = load_afrag(&As[wm * 32 + mt * 16 + r][0], hi);
#pragma unroll
    for (int nt = 0; nt < 2; ++nt) bf[nt] = load_bfrag(&Bs[wn * 32 + nt * 16 + r][0], hi);
#pragma unroll
    for (int mt = 0; mt < 2; ++mt)
#pragma unroll
      for (int nt = 0; nt < 2; ++nt)
        acc[mt][nt] = __builtin_amdgcn_wmma_f32_16x16x32_f16(
            false, af[mt], false, bf[nt], (short)0, acc[mt][nt], false, false);
    __syncthreads();
  }

#pragma unroll
  for (int mt = 0; mt < 2; ++mt)
#pragma unroll
    for (int nt = 0; nt < 2; ++nt)
#pragma unroll
      for (int i = 0; i < 8; ++i) {
        int grow = bm * 64 + wm * 32 + mt * 16 + i + 8 * hi;  // C/D layout: M = vgpr + 8*hi
        int gcol = bn * 128 + wn * 32 + nt * 16 + r;          //             N = lane&15
        float va = acc[mt][nt][i];
        if (EPI == 0) {
          float vv = va + (bias ? bias[gcol] : 0.f);
          outH[(size_t)grow * N + gcol] = (f16)(vv * scale);
        } else if (EPI == 1) {
          int wg = grow >> 6, p = grow & 63;
          int b = wg >> 8, wi = wg & 255;
          int h2 = ((wi >> 4) << 3) + (p >> 3);
          int w2 = ((wi & 15) << 3) + (p & 7);
          int h = (h2 + SHIFT_) & (H_ - 1);   // roll(+SHIFT): dst = win coord + SHIFT
          int w = (w2 + SHIFT_) & (W_ - 1);
          size_t tok = ((size_t)b << 14) + (size_t)h * W_ + w;
          outF[tok * C_ + gcol] = aux[tok * C_ + gcol] + va + bias[gcol];
        } else {  // EPI == 2
          size_t o = (size_t)grow * N + gcol;
          outF[o] = aux[o] + gelu_f(va);
        }
      }
}

// -------- 3) Window attention: one wave per (window, head) --------
__global__ void attention_kernel(const f16* __restrict__ q,
                                 const f16* __restrict__ kv,
                                 f16* __restrict__ o) {
  __shared__ f16 Ps[4][64][72];
  int lane = threadIdx.x & 31, wave = threadIdx.x >> 5;
  int gw = blockIdx.x * 4 + wave;
  int win = gw >> 3, head = gw & 7;
  int wi = win & 255;
  int r = lane & 15, hi = lane >> 4;
  size_t qbase  = (size_t)win * 64 * C_  + head * HD_;
  size_t kvbase = (size_t)win * 64 * 512 + head * HD_;

  // K as B-fragments (4 N-tiles of 16): per lane one contiguous 32B run
  v16h kf[4];
#pragma unroll
  for (int nt = 0; nt < 4; ++nt)
    kf[nt] = *(const v16h*)(kv + kvbase + (size_t)(nt * 16 + r) * 512 + hi * 16);

  // precompute column regions for the mask
  int rcol[4];
#pragma unroll
  for (int nt = 0; nt < 4; ++nt) rcol[nt] = swin_region(wi, nt * 16 + r);

#pragma unroll
  for (int mt = 0; mt < 4; ++mt) {
    v16h qa = load_afrag(q + qbase + (size_t)(mt * 16 + r) * C_, hi);
    v8f acc[4];
#pragma unroll
    for (int nt = 0; nt < 4; ++nt)
#pragma unroll
      for (int e = 0; e < 8; ++e) acc[nt][e] = 0.f;
#pragma unroll
    for (int nt = 0; nt < 4; ++nt)
      acc[nt] = __builtin_amdgcn_wmma_f32_16x16x32_f16(
          false, qa, false, kf[nt], (short)0, acc[nt], false, false);
    // shift mask, branchless (v_cndmask)
#pragma unroll
    for (int i = 0; i < 8; ++i) {
      int rr = swin_region(wi, mt * 16 + i + 8 * hi);
#pragma unroll
      for (int nt = 0; nt < 4; ++nt)
        acc[nt][i] += (rr == rcol[nt]) ? 0.0f : -100.0f;
    }
    // rowwise softmax: row lives in one 16-lane half, spread across 4 nt-tiles
#pragma unroll
    for (int i = 0; i < 8; ++i) {
      float mx = acc[0][i];
#pragma unroll
      for (int nt = 1; nt < 4; ++nt) mx = fmaxf(mx, acc[nt][i]);
      mx = fmaxf(mx, xor_swz<1>(mx));
      mx = fmaxf(mx, xor_swz<2>(mx));
      mx = fmaxf(mx, xor_swz<4>(mx));
      mx = fmaxf(mx, xor_swz<8>(mx));
      float sum = 0.f;
#pragma unroll
      for (int nt = 0; nt < 4; ++nt) { float e = __expf(acc[nt][i] - mx); acc[nt][i] = e; sum += e; }
      sum += xor_swz<1>(sum);
      sum += xor_swz<2>(sum);
      sum += xor_swz<4>(sum);
      sum += xor_swz<8>(sum);
      float invs = 1.0f / sum;
#pragma unroll
      for (int nt = 0; nt < 4; ++nt) acc[nt][i] *= invs;
    }
#pragma unroll
    for (int nt = 0; nt < 4; ++nt)
#pragma unroll
      for (int i = 0; i < 8; ++i)
        Ps[wave][mt * 16 + i + 8 * hi][nt * 16 + r] = (f16)acc[nt][i];
  }
  __syncthreads();

  // V as B-fragments: 2 K-steps x 2 D-tiles (strided per-element gathers)
  v16h vf[2][2];
#pragma unroll
  for (int kt = 0; kt < 2; ++kt)
#pragma unroll
    for (int dt = 0; dt < 2; ++dt)
#pragma unroll
      for (int i = 0; i < 16; ++i)
        vf[kt][dt][i] = kv[kvbase + 256 + (size_t)(kt * 32 + hi * 16 + i) * 512 + dt * 16 + r];

#pragma unroll
  for (int mt = 0; mt < 4; ++mt) {
    v8f oacc[2];
#pragma unroll
    for (int dt = 0; dt < 2; ++dt)
#pragma unroll
      for (int e = 0; e < 8; ++e) oacc[dt][e] = 0.f;
#pragma unroll
    for (int kt = 0; kt < 2; ++kt) {
      const f16* prow = &Ps[wave][mt * 16 + r][0];
      v16h pa = cat8(*(const v8h*)(prow + kt * 32 + hi * 16),
                     *(const v8h*)(prow + kt * 32 + hi * 16 + 8));
#pragma unroll
      for (int dt = 0; dt < 2; ++dt)
        oacc[dt] = __builtin_amdgcn_wmma_f32_16x16x32_f16(
            false, pa, false, vf[kt][dt], (short)0, oacc[dt], false, false);
    }
#pragma unroll
    for (int dt = 0; dt < 2; ++dt)
#pragma unroll
      for (int i = 0; i < 8; ++i) {
        int row = mt * 16 + i + 8 * hi;
        int ch = head * HD_ + dt * 16 + r;
        o[((size_t)win * 64 + row) * C_ + ch] = (f16)oacc[dt][i];
      }
  }
}

// -------- 4) depthwise 3x3 'SAME' on NHWC f16 --------
__global__ void dwconv_kernel(const f16* __restrict__ y1, const float* __restrict__ wdw,
                              f16* __restrict__ yd) {
  int cb = blockIdx.x & 3;
  int tok = blockIdx.x >> 2;
  int c = cb * 256 + threadIdx.x;
  int b = tok >> 14, hw = tok & 16383;
  int h = hw >> 7, w = hw & 127;
  float s = 0.f;
#pragma unroll
  for (int dh = -1; dh <= 1; ++dh) {
    int hh = h + dh;
    if (hh < 0 || hh >= H_) continue;
#pragma unroll
    for (int dw = -1; dw <= 1; ++dw) {
      int ww = w + dw;
      if (ww < 0 || ww >= W_) continue;
      s += (float)y1[(((size_t)b << 14) + (size_t)hh * W_ + ww) * HID_ + c]
           * wdw[c * 9 + (dh + 1) * 3 + (dw + 1)];
    }
  }
  yd[(size_t)tok * HID_ + c] = (f16)s;
}

// -------- 5) ECA: global average pool per (b,c) --------
__global__ void eca_pool_kernel(const f16* __restrict__ y2, float* __restrict__ pooled) {
  __shared__ float red[256];
  int bc = blockIdx.x;                   // b*1024 + c
  int b = bc >> 10, c = bc & 1023;
  float s = 0.f;
  for (int t = threadIdx.x; t < 16384; t += 256)
    s += (float)y2[(((size_t)b << 14) + t) * HID_ + c];
  red[threadIdx.x] = s;
  __syncthreads();
  for (int st = 128; st > 0; st >>= 1) {
    if (threadIdx.x < st) red[threadIdx.x] += red[threadIdx.x + st];
    __syncthreads();
  }
  if (threadIdx.x == 0) pooled[bc] = red[0] * (1.0f / 16384.0f);
}

// -------- 6) ECA conv1d(k=5, pad=2) + sigmoid --------
__global__ void eca_gate_kernel(const float* __restrict__ pooled, const float* __restrict__ weca,
                                float* __restrict__ gate) {
  int id = blockIdx.x * 256 + threadIdx.x;     // 4096 = B*HID
  int b = id >> 10, c = id & 1023;
  float s = 0.f;
#pragma unroll
  for (int j = 0; j < 5; ++j) {
    int cc = c - 2 + j;
    if (cc >= 0 && cc < HID_) s += weca[j] * pooled[b * HID_ + cc];
  }
  gate[id] = 1.0f / (1.0f + __expf(-s));
}

// -------- 7) channel gate + split + gelu gating -> y4 (tok x 512) --------
__global__ void gate_gelu_kernel(const f16* __restrict__ y2, const float* __restrict__ gate,
                                 f16* __restrict__ y4) {
  size_t idx = (size_t)blockIdx.x * 256 + threadIdx.x;   // TOK_*512
  size_t m = idx >> 9; int j = (int)(idx & 511);
  int b = (int)(m >> 14);
  float g = (float)y2[m * HID_ + j]       * gate[b * HID_ + j];
  float h = (float)y2[m * HID_ + 512 + j] * gate[b * HID_ + 512 + j];
  y4[m * 512 + j] = (f16)(gelu_f(g) * h);
}

extern "C" void kernel_launch(void* const* d_in, const int* in_sizes, int n_in,
                              void* d_out, int out_size, void* d_ws, size_t ws_size,
                              hipStream_t stream) {
  (void)in_sizes; (void)n_in; (void)out_size; (void)ws_size;
  const float* x     = (const float*)d_in[0];
  const float* g1    = (const float*)d_in[1];
  const float* b1    = (const float*)d_in[2];
  const float* wq    = (const float*)d_in[3];
  const float* bq    = (const float*)d_in[4];
  const float* wkv   = (const float*)d_in[5];
  const float* bkv   = (const float*)d_in[6];
  const float* wproj = (const float*)d_in[7];
  const float* bproj = (const float*)d_in[8];
  const float* w_in  = (const float*)d_in[9];
  const float* w_dw  = (const float*)d_in[10];
  const float* w_pw  = (const float*)d_in[11];
  const float* w_eca = (const float*)d_in[12];
  const float* w_out = (const float*)d_in[13];
  float* out = (float*)d_out;

  char* ws = (char*)d_ws;
  const size_t MB = 1ull << 20;
  // Phase-aliased workspace regions (peak ~448MB)
  f16*   xw     = (f16*)(ws + 0);             // 32MB  R1: LN'd windows
  f16*   y1     = (f16*)(ws + 0);             // 128MB R1 (after xw dead)
  f16*   qb     = (f16*)(ws + 128 * MB);      // 32MB  R2
  f16*   kvb    = (f16*)(ws + 160 * MB);      // 64MB  R2
  f16*   yd     = (f16*)(ws + 128 * MB);      // 128MB R2 (after q/kv dead)
  f16*   y4     = (f16*)(ws + 128 * MB);      // 64MB  R2 (after yd dead)
  f16*   attn   = (f16*)(ws + 256 * MB);      // 32MB  R3
  f16*   y2     = (f16*)(ws + 256 * MB);      // 128MB R3 (after attn dead)
  float* x2     = (float*)(ws + 384 * MB);    // 64MB  R4 (persistent)
  float* pooled = (float*)(ws + 448 * MB);
  float* gate   = (float*)(ws + 448 * MB + 64 * 1024);

  // 1) LN + shift + window partition
  ln_window_kernel<<<8192, 256, 0, stream>>>(x, g1, b1, xw);
  // 2) Q = (xw@wq^T + bq) * hd^-0.5   (M=65536,K=256,N=256)
  gemm_wmma_kernel<f16, 0><<<1024 * 2, 256, 0, stream>>>(
      xw, wq, bq, qb, nullptr, nullptr, TOK_, C_, C_, 0.17677669529663687f);
  // 3) KV = xw@wkv^T + bkv            (N=512)
  gemm_wmma_kernel<f16, 0><<<1024 * 4, 256, 0, stream>>>(
      xw, wkv, bkv, kvb, nullptr, nullptr, TOK_, 512, C_, 1.0f);
  // 4) window attention (8192 window-head waves)
  attention_kernel<<<2048, 128, 0, stream>>>(qb, kvb, attn);
  // 5) proj + window-reverse + unshift + residual -> x2
  gemm_wmma_kernel<f16, 1><<<1024 * 2, 256, 0, stream>>>(
      attn, wproj, bproj, nullptr, x2, x, TOK_, C_, C_, 1.0f);
  // 6) y1 = x2 @ w_in^T               (N=1024)
  gemm_wmma_kernel<float, 0><<<1024 * 8, 256, 0, stream>>>(
      x2, w_in, nullptr, y1, nullptr, nullptr, TOK_, HID_, C_, 1.0f);
  // 7) depthwise 3x3
  dwconv_kernel<<<TOK_ * 4, 256, 0, stream>>>(y1, w_dw, yd);
  // 8) y2 = yd @ w_pw^T               (K=1024,N=1024 — the 137 GFLOP GEMM)
  gemm_wmma_kernel<f16, 0><<<1024 * 8, 256, 0, stream>>>(
      yd, w_pw, nullptr, y2, nullptr, nullptr, TOK_, HID_, HID_, 1.0f);
  // 9) ECA
  eca_pool_kernel<<<4096, 256, 0, stream>>>(y2, pooled);
  eca_gate_kernel<<<16, 256, 0, stream>>>(pooled, w_eca, gate);
  // 10) gate + split + gelu gating -> y4
  gate_gelu_kernel<<<131072, 256, 0, stream>>>(y2, gate, y4);
  // 11) out = x2 + gelu(y4 @ w_out^T) (K=512,N=256)
  gemm_wmma_kernel<f16, 2><<<1024 * 2, 256, 0, stream>>>(
      y4, w_out, nullptr, nullptr, out, x2, TOK_, C_, 512, 1.0f);
}
